// MinGRU_33526514713036
// MI455X (gfx1250) — matrix operand order repaired and checked by reading, not compiled, not run on hardware
//
#include <hip/hip_runtime.h>
#include <hip/hip_bf16.h>
#include <stddef.h>
#include <stdint.h>

// ---------------------------------------------------------------------------
// MinGRU (2 layers) for MI455X / gfx1250.
//   split  : fp32 -> bf16 hi + bf16 residual (one-time, halves GEMM traffic)
//   gemm   : split-bf16 WMMA (v_wmma_f32_16x16x32_bf16, 3 terms ~ fp32 acc),
//            double-buffered LDS fed by GLOBAL_LOAD_ASYNC_TO_LDS_B128
//   scan   : h_t = (1-z)h_{t-1} + z g(x_t)  (linear space; all terms positive)
// Dims fixed by the reference: B=8, S=4096, D=H=1024.
// ---------------------------------------------------------------------------

#define B_DIM 8
#define S_DIM 4096
#define H_DIM 1024
#define K_DIM 1024            // Din for both layers
#define N_DIM 2048            // 2H
#define M_DIM (B_DIM * S_DIM) // 32768

typedef __attribute__((ext_vector_type(16))) __bf16 v16bf;
typedef __attribute__((ext_vector_type(8)))  __bf16 v8bf;
typedef __attribute__((ext_vector_type(4)))  __bf16 v4bf;
typedef __attribute__((ext_vector_type(8)))  float  v8f;
typedef int b128i __attribute__((vector_size(16)));   // builtin's B128 unit

#define AS1 __attribute__((address_space(1)))
#define AS3 __attribute__((address_space(3)))

// ---- async global->LDS 16-byte copy (ASYNCcnt-tracked) --------------------
__device__ __forceinline__ void async_b128(const __bf16* g, __bf16* l) {
#if __has_builtin(__builtin_amdgcn_global_load_async_to_lds_b128)
  __builtin_amdgcn_global_load_async_to_lds_b128(
      (AS1 b128i*)g, (AS3 b128i*)l, 0, 0);
#else
  const unsigned lds = (unsigned)(uintptr_t)(AS3 void*)l;
  asm volatile("global_load_async_to_lds_b128 %0, %1, off"
               :: "v"(lds), "v"((unsigned long long)(uintptr_t)g) : "memory");
#endif
}

#if __has_builtin(__builtin_amdgcn_s_wait_asynccnt)
#define WAIT_ASYNC(n) __builtin_amdgcn_s_wait_asynccnt(n)
#else
#define WAIT_ASYNC(n) asm volatile("s_wait_asynccnt %0" :: "i"(n) : "memory")
#endif

// ---------------------------------------------------------------------------
// One-time fp32 -> (bf16 hi, bf16 lo) split. hi+lo reproduces x to ~2^-17 rel.
// ---------------------------------------------------------------------------
__global__ __launch_bounds__(256)
void split_f32_bf16(const float* __restrict__ in,
                    __bf16* __restrict__ hi, __bf16* __restrict__ lo) {
  const size_t i = ((size_t)blockIdx.x * 256 + threadIdx.x) * 4;
  const float4 v4 = *(const float4*)(in + i);
  const float f[4] = {v4.x, v4.y, v4.z, v4.w};
  v4bf h, l;
#pragma unroll
  for (int j = 0; j < 4; ++j) {
    const __bf16 hh = (__bf16)f[j];
    h[j] = hh;
    l[j] = (__bf16)(f[j] - (float)hh);
  }
  *(v4bf*)(hi + i) = h;
  *(v4bf*)(lo + i) = l;
}

// ---------------------------------------------------------------------------
// GEMM: C(MxN) = A @ B^T + bias. A,B pre-split bf16 (hi,lo), row-major over K.
// Block 128x128 (8 waves), wave tile 64x32 = 4x2 frags, K-step 32.
// 3 WMMAs per tile: hi*hi + hi*lo + lo*hi.
// ---------------------------------------------------------------------------
#define TM 128
#define TN 128
#define KT 32
#define LDK2 40  // LDS row stride in halves: 80B (16B aligned, 20 dwords)

__global__ __launch_bounds__(256, 1)
void mingru_gemm_bf16x3(const __bf16* __restrict__ Ahg, const __bf16* __restrict__ Alg,
                        const __bf16* __restrict__ Bhg, const __bf16* __restrict__ Blg,
                        const float*  __restrict__ bias,
                        float* __restrict__ C)
{
  __shared__ __bf16 sAh[2][TM][LDK2];
  __shared__ __bf16 sAl[2][TM][LDK2];
  __shared__ __bf16 sBh[2][TN][LDK2];
  __shared__ __bf16 sBl[2][TN][LDK2];

  const int tid  = threadIdx.x;
  const int lane = tid & 31;
  const int w    = tid >> 5;      // wave 0..7
  const int wm   = w >> 2;        // 0..1  (M dir)
  const int wn   = w & 3;         // 0..3  (N dir)
  const int l15  = lane & 15;
  const int hi16 = lane >> 4;     // lane group 0/1

  const int mblk = blockIdx.y * TM;
  const int nblk = blockIdx.x * TN;

  // Stage one K-slice (tile) of A/B hi+lo into LDS buffer `buf` via async DMA.
  // 512 b128 per array per tile -> 2 per thread per array (8 asyncs/thread).
  auto stage = [&](int buf, int kb) {
#pragma unroll
    for (int i = 0; i < 2; ++i) {
      const int flat = tid + i * 256;        // 0..511
      const int r    = flat >> 2;            // tile row 0..127
      const int qh   = (flat & 3) * 8;       // half offset 0,8,16,24 (16B units)
      const size_t ga = (size_t)(mblk + r) * K_DIM + kb + qh;
      const size_t gb = (size_t)(nblk + r) * K_DIM + kb + qh;
      async_b128(Ahg + ga, &sAh[buf][r][qh]);
      async_b128(Alg + ga, &sAl[buf][r][qh]);
      async_b128(Bhg + gb, &sBh[buf][r][qh]);
      async_b128(Blg + gb, &sBl[buf][r][qh]);
    }
  };

  v8f acc[4][2] = {};

  constexpr int T = K_DIM / KT;  // 32 K-tiles
  stage(0, 0);

  for (int t = 0; t < T; ++t) {
    const int cur = t & 1;
    if (t + 1 < T) {
      stage(cur ^ 1, (t + 1) * KT);
      WAIT_ASYNC(8);   // previous stage's 8 copies done; next 8 in flight
    } else {
      WAIT_ASYNC(0);
    }
    __syncthreads();   // all waves' current tile resident in LDS

    // ---- WMMA fragments from LDS (ISA bf16 lane layouts) ----
    // A 16x32: lanes 0-15 K {0..7,16..23}; lanes 16-31 K {8..15,24..31}
    v16bf afh[4], afl[4];
    const int ka = hi16 ? 8 : 0;
#pragma unroll
    for (int mi = 0; mi < 4; ++mi) {
      const int r = wm * 64 + mi * 16 + l15;
      v8bf h0 = *(const v8bf*)&sAh[cur][r][ka];
      v8bf h1 = *(const v8bf*)&sAh[cur][r][ka + 16];
      afh[mi] = __builtin_shufflevector(h0, h1, 0,1,2,3,4,5,6,7,8,9,10,11,12,13,14,15);
      v8bf l0 = *(const v8bf*)&sAl[cur][r][ka];
      v8bf l1 = *(const v8bf*)&sAl[cur][r][ka + 16];
      afl[mi] = __builtin_shufflevector(l0, l1, 0,1,2,3,4,5,6,7,8,9,10,11,12,13,14,15);
    }
    // B 32x16: lanes 0-15 K {0..7,16..23}; lanes 16-31 K {8..15,24..31}
    v16bf bfh[2], bfl[2];
    const int kbq = hi16 ? 8 : 0;
#pragma unroll
    for (int ni = 0; ni < 2; ++ni) {
      const int r = wn * 32 + ni * 16 + l15;
      v8bf h0 = *(const v8bf*)&sBh[cur][r][kbq];
      v8bf h1 = *(const v8bf*)&sBh[cur][r][kbq + 16];
      bfh[ni] = __builtin_shufflevector(h0, h1, 0,1,2,3,4,5,6,7,8,9,10,11,12,13,14,15);
      v8bf l0 = *(const v8bf*)&sBl[cur][r][kbq];
      v8bf l1 = *(const v8bf*)&sBl[cur][r][kbq + 16];
      bfl[ni] = __builtin_shufflevector(l0, l1, 0,1,2,3,4,5,6,7,8,9,10,11,12,13,14,15);
    }

    // ---- 4x2 tiles x 3 split terms = 24 WMMAs per K-step per wave ----
#pragma unroll
    for (int mi = 0; mi < 4; ++mi) {
#pragma unroll
      for (int ni = 0; ni < 2; ++ni) {
        acc[mi][ni] = __builtin_amdgcn_wmma_f32_16x16x32_bf16(
            false, afh[mi], false, bfh[ni], (short)0, acc[mi][ni], false, false);
        acc[mi][ni] = __builtin_amdgcn_wmma_f32_16x16x32_bf16(
            false, afh[mi], false, bfl[ni], (short)0, acc[mi][ni], false, false);
        acc[mi][ni] = __builtin_amdgcn_wmma_f32_16x16x32_bf16(
            false, afl[mi], false, bfh[ni], (short)0, acc[mi][ni], false, false);
      }
    }
    __syncthreads();   // everyone done reading buf[cur] before it is restaged
  }

  // ---- epilogue: fused bias, store fp32. C/D: VGPR r -> M=r+8*hi16, N=l15 --
#pragma unroll
  for (int ni = 0; ni < 2; ++ni) {
    const int n  = nblk + wn * 32 + ni * 16 + l15;
    const float bb = bias[n];
#pragma unroll
    for (int mi = 0; mi < 4; ++mi) {
#pragma unroll
      for (int r = 0; r < 8; ++r) {
        const int m = mblk + wm * 64 + mi * 16 + r + hi16 * 8;
        C[(size_t)m * N_DIM + n] = acc[mi][ni][r] + bb;
      }
    }
  }
}

// ---------------------------------------------------------------------------
// Scan. SPLIT=true  -> emit h as bf16 hi/lo (feeds layer-1 GEMM directly)
//       SPLIT=false -> emit fp32 (final output)
// One thread per (b,h) channel; coalesced over h each step; bandwidth bound.
// ---------------------------------------------------------------------------
template <bool SPLIT>
__global__ __launch_bounds__(256)
void mingru_scan(const float* __restrict__ gh,   // (B, S, 2H)
                 float*  __restrict__ of32,      // (B, S, H) or null
                 __bf16* __restrict__ ohi,       // (B, S, H) or null
                 __bf16* __restrict__ olo)
{
  const int ch = blockIdx.x * blockDim.x + threadIdx.x;   // 0..B*H-1
  const int b  = ch >> 10;          // / H
  const int h  = ch & (H_DIM - 1);

  const float* gate = gh + (size_t)b * S_DIM * N_DIM + h;
  const float* hid  = gate + H_DIM;
  const size_t obase = (size_t)b * S_DIM * H_DIM + h;

  float hv = 0.5f;
#pragma unroll 4
  for (int s = 0; s < S_DIM; ++s) {
    const float g = gate[(size_t)s * N_DIM];
    const float x = hid [(size_t)s * N_DIM];
    const float z  = 1.0f / (1.0f + __expf(-g));
    const float gg = (x >= 0.0f) ? (x + 0.5f) : 1.0f / (1.0f + __expf(-x));
    hv = hv + z * (gg - hv);
    const size_t oi = obase + (size_t)s * H_DIM;
    if (SPLIT) {
      const __bf16 hh = (__bf16)hv;
      ohi[oi] = hh;
      olo[oi] = (__bf16)(hv - (float)hh);
    } else {
      of32[oi] = hv;
    }
  }
}

// ---------------------------------------------------------------------------
extern "C" void kernel_launch(void* const* d_in, const int* in_sizes, int n_in,
                              void* d_out, int out_size, void* d_ws, size_t ws_size,
                              hipStream_t stream) {
  const float* x  = (const float*)d_in[0];  // (B,S,D)
  const float* W0 = (const float*)d_in[1];  // (2H,D)
  const float* b0 = (const float*)d_in[2];  // (2H)
  const float* W1 = (const float*)d_in[3];  // (2H,H)
  const float* b1 = (const float*)d_in[4];  // (2H)

  // Workspace layout (400 MiB total):
  const size_t ACT_N = (size_t)M_DIM * K_DIM;   // 33,554,432
  const size_t W_N   = (size_t)N_DIM * K_DIM;   //  2,097,152
  float*  GH   = (float*)d_ws;                         // (M,2H) 256 MiB
  __bf16* ACTh = (__bf16*)(GH + (size_t)M_DIM * N_DIM);// 64 MiB
  __bf16* ACTl = ACTh + ACT_N;                         // 64 MiB
  __bf16* W0h  = ACTl + ACT_N;                         //  4 MiB
  __bf16* W0l  = W0h + W_N;
  __bf16* W1h  = W0l + W_N;
  __bf16* W1l  = W1h + W_N;

  const dim3 blk(256);
  const dim3 ggrd(N_DIM / TN, M_DIM / TM);             // 16 x 256 blocks
  const dim3 sgrd((B_DIM * H_DIM) / 256);              // 32 blocks

  // One-time precision splits
  split_f32_bf16<<<dim3(ACT_N / 4 / 256), blk, 0, stream>>>(x,  ACTh, ACTl);
  split_f32_bf16<<<dim3(W_N   / 4 / 256), blk, 0, stream>>>(W0, W0h,  W0l);
  split_f32_bf16<<<dim3(W_N   / 4 / 256), blk, 0, stream>>>(W1, W1h,  W1l);

  // Layer 0
  mingru_gemm_bf16x3<<<ggrd, blk, 0, stream>>>(ACTh, ACTl, W0h, W0l, b0, GH);
  mingru_scan<true><<<sgrd, blk, 0, stream>>>(GH, nullptr, ACTh, ACTl); // h1 -> splits
  // Layer 1
  mingru_gemm_bf16x3<<<ggrd, blk, 0, stream>>>(ACTh, ACTl, W1h, W1l, b1, GH);
  mingru_scan<false><<<sgrd, blk, 0, stream>>>(GH, (float*)d_out, nullptr, nullptr);
}
